// GraphPropLayer_37185826849402
// MI455X (gfx1250) — matrix operand order repaired and checked by reading, not compiled, not run on hardware
//
#include <hip/hip_runtime.h>

// ---------------------------------------------------------------------------
// GraphPropLayer on MI455X (gfx1250, wave32, WMMA bf16 -> f32)
//
// Edge MLPs dominate (~2.7e11 FLOP): bf16 WMMA with f32 accumulation.
// Each wave now owns TWO 16-edge A tiles so every B (weight) fragment feeds
// two v_wmma ops -> halves L2 weight traffic per FLOP vs round 1.
// ---------------------------------------------------------------------------

#define NN     50000
#define NE     512000
#define DD     128
#define HH     256

typedef __attribute__((ext_vector_type(16))) __bf16 v16bf;
typedef __attribute__((ext_vector_type(8)))  __bf16 v8bf;
typedef __attribute__((ext_vector_type(8)))  float  v8f;

// bf16 weight pool offsets (elements)
#define OFF_WF1 0
#define OFF_WF2 65536
#define OFF_WR1 131072
#define OFF_WR2 196608
#define OFF_WIH 262144          // 384*256
#define OFF_WHH 360448          // 384*128
#define W_TOTAL 409600
#define AGG_BYTES ((size_t)NN * HH * sizeof(float))   // 51.2 MB in d_ws

// ---------------------------------------------------------------------------
__global__ __launch_bounds__(256) void prep_weights(
    const float* __restrict__ Wf1, const float* __restrict__ Wf2,
    const float* __restrict__ Wr1, const float* __restrict__ Wr2,
    const float* __restrict__ Wih, const float* __restrict__ Whh,
    __bf16* __restrict__ out) {
  for (int i = blockIdx.x * blockDim.x + threadIdx.x; i < W_TOTAL;
       i += gridDim.x * blockDim.x) {
    float v;
    if      (i < OFF_WF2) v = Wf1[i];
    else if (i < OFF_WR1) v = Wf2[i - OFF_WF2];
    else if (i < OFF_WR2) v = Wr1[i - OFF_WR1];
    else if (i < OFF_WIH) v = Wr2[i - OFF_WR2];
    else if (i < OFF_WHH) v = Wih[i - OFF_WIH];
    else                  v = Whh[i - OFF_WHH];
    out[i] = (__bf16)v;
  }
}

// --- fragment loaders -------------------------------------------------------
// A frag (16x32 bf16) from LDS row-major [rows][256], with optional 128-col
// rotation (reverse-net input = forward input rotated by 128 columns).
// ISA layout: lane<16 holds K k0..k0+7 and k0+16..k0+23; lane>=16 is +8.
__device__ __forceinline__ v16bf load_a256(const __bf16* base, int row, int kt,
                                           int rot, int lane) {
  int k0 = kt * 32 + ((lane & 16) ? 8 : 0);
  const __bf16* rp = base + row * HH;
  v8bf lo = *(const v8bf*)(rp + ((k0 + rot) & 255));
  v8bf hi = *(const v8bf*)(rp + ((k0 + 16 + rot) & 255));
  return __builtin_shufflevector(lo, hi, 0, 1, 2, 3, 4, 5, 6, 7, 8, 9, 10, 11,
                                 12, 13, 14, 15);
}
// A frag from LDS row-major [rows][128] (GRU hidden-state operand, K=128).
__device__ __forceinline__ v16bf load_a128(const __bf16* base, int row, int kt,
                                           int lane) {
  int k0 = kt * 32 + ((lane & 16) ? 8 : 0);
  const __bf16* rp = base + row * DD;
  v8bf lo = *(const v8bf*)(rp + k0);
  v8bf hi = *(const v8bf*)(rp + k0 + 16);
  return __builtin_shufflevector(lo, hi, 0, 1, 2, 3, 4, 5, 6, 7, 8, 9, 10, 11,
                                 12, 13, 14, 15);
}
// B frag (32x16): lane = output column; lane<16 holds K k0..k0+7 and
// k0+16..k0+23, lane>=16 is +8 (matches A). Two 16B loads per lane.
__device__ __forceinline__ v16bf load_b(const __bf16* W, int lane, int nt,
                                        int kt, int ldk) {
  int o  = nt * 16 + (lane & 15);
  int k0 = kt * 32 + ((lane & 16) ? 8 : 0);
  const __bf16* rp = W + (size_t)o * ldk;
  v8bf lo = *(const v8bf*)(rp + k0);
  v8bf hi = *(const v8bf*)(rp + k0 + 16);
  return __builtin_shufflevector(lo, hi, 0, 1, 2, 3, 4, 5, 6, 7, 8, 9, 10, 11,
                                 12, 13, 14, 15);
}

__device__ __forceinline__ v8f wmma_bf16(v16bf a, v16bf b, v8f c) {
  return __builtin_amdgcn_wmma_f32_16x16x32_bf16(false, a, false, b, (short)0,
                                                 c, false, false);
}

__device__ __forceinline__ float sigmoidf_(float x) {
  return 1.0f / (1.0f + __expf(-x));
}

// ---------------------------------------------------------------------------
// Edge kernel: 2 waves/block, 32 edges/wave (two 16-row tiles sharing each
// B fragment), 64 edges/block, 8000 blocks. LDS 64 KB + index staging.
// ---------------------------------------------------------------------------
__global__ __launch_bounds__(64) void edge_kernel(
    const float* __restrict__ node_states,
    const int* __restrict__ from_idx, const int* __restrict__ to_idx,
    const __bf16* __restrict__ wbf,
    const float* __restrict__ bf1, const float* __restrict__ bf2,
    const float* __restrict__ br1, const float* __restrict__ br2,
    float* __restrict__ agg) {
  __shared__ __bf16 sX[64 * HH];   // gathered concat inputs, 32 KB
  __shared__ __bf16 sH[64 * HH];   // layer-1 activations,    32 KB
  __shared__ int    sFrom[64];
  __shared__ int    sTo[64];

  const int lane  = threadIdx.x & 31;
  const int wave  = threadIdx.x >> 5;
  const int ebase = blockIdx.x * 64;

  // stage the 64 edge index pairs once
  sFrom[threadIdx.x] = from_idx[ebase + threadIdx.x];
  sTo[threadIdx.x]   = to_idx[ebase + threadIdx.x];
  __syncthreads();

  // ---- cooperative gather: concat(s[from], s[to]) -> bf16 LDS -------------
  for (int i = threadIdx.x; i < 64 * 64; i += 64) {
    int e  = i >> 6;
    int ch = i & 63;
    int sn = (ch < 32) ? sFrom[e] : sTo[e];
    int c4 = ch & 31;
    const float4 f =
        *(const float4*)(node_states + (size_t)sn * DD + c4 * 4);
    int col = ((ch < 32) ? 0 : DD) + c4 * 4;
    __bf16* d = &sX[e * HH + col];
    d[0] = (__bf16)f.x; d[1] = (__bf16)f.y;
    d[2] = (__bf16)f.z; d[3] = (__bf16)f.w;
  }
  __syncthreads();

  const int wrow  = wave * 32;             // this wave's 32-row LDS block
  const int arow0 = wrow + (lane & 15);    // A-fragment rows, tile 0 / tile 1
  const int arow1 = arow0 + 16;
  const int colL  = lane & 15;

  for (int net = 0; net < 2; ++net) {
    const __bf16* W1 = wbf + (net ? OFF_WR1 : OFF_WF1);
    const __bf16* W2 = wbf + (net ? OFF_WR2 : OFF_WF2);
    const float*  b1 = net ? br1 : bf1;
    const float*  b2 = net ? br2 : bf2;
    const int*  didx = net ? sFrom : sTo;
    const int    rot = net ? 128 : 0;

    // ---- layer 1: h = relu(X @ W1^T + b1), 32x256 per wave ----------------
    for (int nt = 0; nt < 16; ++nt) {
      v8f acc0 = {}, acc1 = {};
#pragma unroll
      for (int kt = 0; kt < 8; ++kt) {
        v16bf b  = load_b(W1, lane, nt, kt, HH);          // shared by 2 WMMAs
        v16bf a0 = load_a256(sX, arow0, kt, rot, lane);
        v16bf a1 = load_a256(sX, arow1, kt, rot, lane);
        acc0 = wmma_bf16(a0, b, acc0);
        acc1 = wmma_bf16(a1, b, acc1);
      }
      const int  colc = nt * 16 + colL;
      const float bias = b1[colc];
#pragma unroll
      for (int r = 0; r < 8; ++r) {
        int rr = (lane & 16) ? 8 + r : r;
        float v0 = acc0[r] + bias;
        float v1 = acc1[r] + bias;
        sH[(wrow + rr) * HH + colc]      = (__bf16)(v0 > 0.0f ? v0 : 0.0f);
        sH[(wrow + 16 + rr) * HH + colc] = (__bf16)(v1 > 0.0f ? v1 : 0.0f);
      }
    }
    __syncthreads();   // order LDS writes vs cross-lane A-fragment reads

    // ---- layer 2 + scatter-add into agg -----------------------------------
    int dest0[8], dest1[8];
#pragma unroll
    for (int r = 0; r < 8; ++r) {
      int rr   = (lane & 16) ? 8 + r : r;
      dest0[r] = didx[wrow + rr];
      dest1[r] = didx[wrow + 16 + rr];
    }
    for (int nt = 0; nt < 16; ++nt) {
      v8f acc0 = {}, acc1 = {};
#pragma unroll
      for (int kt = 0; kt < 8; ++kt) {
        v16bf b  = load_b(W2, lane, nt, kt, HH);
        v16bf a0 = load_a256(sH, arow0, kt, 0, lane);
        v16bf a1 = load_a256(sH, arow1, kt, 0, lane);
        acc0 = wmma_bf16(a0, b, acc0);
        acc1 = wmma_bf16(a1, b, acc1);
      }
      const int  colc = nt * 16 + colL;
      const float bias = b2[colc];
#pragma unroll
      for (int r = 0; r < 8; ++r) {
        unsafeAtomicAdd(&agg[(size_t)dest0[r] * HH + colc], acc0[r] + bias);
        unsafeAtomicAdd(&agg[(size_t)dest1[r] * HH + colc], acc1[r] + bias);
      }
    }
    __syncthreads();   // sH reused by the reverse net next iteration
  }
}

// ---------------------------------------------------------------------------
// GRU kernel: 4 waves/block, 16 nodes/wave, 64 nodes/block, 782 blocks.
// Per column-tile t (8 tiles of 16 cols), compute the six gate accumulators
// with WMMA; the 16x16 C layout keeps (m,c) for all gates in the same lane.
// ---------------------------------------------------------------------------
__global__ __launch_bounds__(128) void gru_kernel(
    const float* __restrict__ agg, const float* __restrict__ node_states,
    const __bf16* __restrict__ wbf,
    const float* __restrict__ b_ih, const float* __restrict__ b_hh,
    float* __restrict__ out) {
  __shared__ __bf16 sXa[64 * HH];  // agg  (x) in bf16, 32 KB
  __shared__ __bf16 sXh[64 * DD];  // h          bf16, 16 KB

  const int lane  = threadIdx.x & 31;
  const int wave  = threadIdx.x >> 5;
  const int nbase = blockIdx.x * 64;

  for (int i = threadIdx.x; i < 64 * 64; i += 128) {
    int rw = i >> 6, c4 = i & 63;
    int node = min(nbase + rw, NN - 1);
    const float4 f = *(const float4*)(agg + (size_t)node * HH + c4 * 4);
    __bf16* d = &sXa[rw * HH + c4 * 4];
    d[0] = (__bf16)f.x; d[1] = (__bf16)f.y;
    d[2] = (__bf16)f.z; d[3] = (__bf16)f.w;
  }
  for (int i = threadIdx.x; i < 64 * 32; i += 128) {
    int rw = i >> 5, c4 = i & 31;
    int node = min(nbase + rw, NN - 1);
    const float4 f = *(const float4*)(node_states + (size_t)node * DD + c4 * 4);
    __bf16* d = &sXh[rw * DD + c4 * 4];
    d[0] = (__bf16)f.x; d[1] = (__bf16)f.y;
    d[2] = (__bf16)f.z; d[3] = (__bf16)f.w;
  }
  __syncthreads();

  const __bf16* Wih = wbf + OFF_WIH;   // [384,256]
  const __bf16* Whh = wbf + OFF_WHH;   // [384,128]
  const int wrow = wave * 16;
  const int arow = wrow + (lane & 15);

  for (int t = 0; t < 8; ++t) {
    v8f gr = {}, gz = {}, gn = {}, hr = {}, hz = {}, hn = {};
#pragma unroll
    for (int kt = 0; kt < 8; ++kt) {               // gi = x @ W_ih^T, K=256
      v16bf a = load_a256(sXa, arow, kt, 0, lane);
      gr = wmma_bf16(a, load_b(Wih, lane, t,      kt, HH), gr);
      gz = wmma_bf16(a, load_b(Wih, lane, t + 8,  kt, HH), gz);
      gn = wmma_bf16(a, load_b(Wih, lane, t + 16, kt, HH), gn);
    }
#pragma unroll
    for (int kt = 0; kt < 4; ++kt) {               // gh = h @ W_hh^T, K=128
      v16bf a = load_a128(sXh, arow, kt, lane);
      hr = wmma_bf16(a, load_b(Whh, lane, t,      kt, DD), hr);
      hz = wmma_bf16(a, load_b(Whh, lane, t + 8,  kt, DD), hz);
      hn = wmma_bf16(a, load_b(Whh, lane, t + 16, kt, DD), hn);
    }
    const int cl = t * 16 + (lane & 15);
    const float bir = b_ih[cl], biz = b_ih[DD + cl], bin = b_ih[2 * DD + cl];
    const float bhr = b_hh[cl], bhz = b_hh[DD + cl], bhn = b_hh[2 * DD + cl];
#pragma unroll
    for (int r = 0; r < 8; ++r) {
      int m    = wrow + ((lane & 16) ? 8 + r : r);
      int node = nbase + m;
      if (node < NN) {
        float rg = sigmoidf_((gr[r] + bir) + (hr[r] + bhr));
        float zg = sigmoidf_((gz[r] + biz) + (hz[r] + bhz));
        float ng = tanhf((gn[r] + bin) + rg * (hn[r] + bhn));
        float h0 = node_states[(size_t)node * DD + cl];
        out[(size_t)node * DD + cl] = (1.0f - zg) * ng + zg * h0;
      }
    }
  }
}

// ---------------------------------------------------------------------------
extern "C" void kernel_launch(void* const* d_in, const int* in_sizes, int n_in,
                              void* d_out, int out_size, void* d_ws,
                              size_t ws_size, hipStream_t stream) {
  const float* node_states = (const float*)d_in[0];
  const int*   from_idx    = (const int*)d_in[1];
  const int*   to_idx      = (const int*)d_in[2];
  const float* Wf1 = (const float*)d_in[3];
  const float* bf1 = (const float*)d_in[4];
  const float* Wf2 = (const float*)d_in[5];
  const float* bf2 = (const float*)d_in[6];
  const float* Wr1 = (const float*)d_in[7];
  const float* br1 = (const float*)d_in[8];
  const float* Wr2 = (const float*)d_in[9];
  const float* br2 = (const float*)d_in[10];
  const float* Wih = (const float*)d_in[11];
  const float* Whh = (const float*)d_in[12];
  const float* bih = (const float*)d_in[13];
  const float* bhh = (const float*)d_in[14];
  float* out = (float*)d_out;

  float*  agg = (float*)d_ws;
  __bf16* wbf = (__bf16*)((char*)d_ws + AGG_BYTES);

  hipMemsetAsync(agg, 0, AGG_BYTES, stream);
  prep_weights<<<(W_TOTAL + 255) / 256, 256, 0, stream>>>(Wf1, Wf2, Wr1, Wr2,
                                                          Wih, Whh, wbf);
  edge_kernel<<<NE / 64, 64, 0, stream>>>(node_states, from_idx, to_idx, wbf,
                                          bf1, bf2, br1, br2, agg);
  gru_kernel<<<(NN + 63) / 64, 128, 0, stream>>>(agg, node_states, wbf, bih,
                                                 bhh, out);
}
